// MultiHeadAttention_90469191123138
// MI455X (gfx1250) — matrix operand order repaired and checked by reading, not compiled, not run on hardware
//
#include <hip/hip_runtime.h>
#include <hip/hip_bf16.h>

// MHA forward for MI455X (gfx1250, wave32, WMMA 16x16x32 bf16, f32 accum).
// B=4, S=2048, D=1024, H=16, DH=64.
// v2: 64x64-tile projection GEMMs (16 WMMA / K-step, ~21 FLOP/B from L2),
//     32-query attention waves sharing K/V fragments, masked work isolated
//     to the single diagonal 32-key block per query group.

#define B_  4
#define S_  2048
#define D_  1024
#define H_  16
#define DH_ 64

typedef __attribute__((ext_vector_type(16))) __bf16 v16bf;
typedef __attribute__((ext_vector_type(8)))  float  v8f;

union BF8 { uint4 u; __bf16 b[8]; };

__device__ inline v8f vzero8() {
  v8f z;
#pragma unroll
  for (int i = 0; i < 8; ++i) z[i] = 0.0f;
  return z;
}

// A-fragment (16x32 bf16, MxK): lane holds row m=lane&15; ka = (lane>=16)?8:0.
// elements 0..15 <- K = 2*ka+0..15 (16 consecutive K)
__device__ inline v16bf loadA_bf16(const __bf16* p, int ka) {
  BF8 lo, hi;
  lo.u = *reinterpret_cast<const uint4*>(p + 2 * ka);
  hi.u = *reinterpret_cast<const uint4*>(p + 2 * ka + 8);
  v16bf r;
#pragma unroll
  for (int i = 0; i < 8; ++i) { r[i] = lo.b[i]; r[i + 8] = hi.b[i]; }
  return r;
}

__device__ inline v16bf loadA_f32(const float* p, int ka) {
  const float* q = p + 2 * ka;
  float4 a0 = *reinterpret_cast<const float4*>(q);
  float4 a1 = *reinterpret_cast<const float4*>(q + 4);
  float4 b0 = *reinterpret_cast<const float4*>(q + 8);
  float4 b1 = *reinterpret_cast<const float4*>(q + 12);
  v16bf r;
  r[0]  = (__bf16)a0.x; r[1]  = (__bf16)a0.y; r[2]  = (__bf16)a0.z; r[3]  = (__bf16)a0.w;
  r[4]  = (__bf16)a1.x; r[5]  = (__bf16)a1.y; r[6]  = (__bf16)a1.z; r[7]  = (__bf16)a1.w;
  r[8]  = (__bf16)b0.x; r[9]  = (__bf16)b0.y; r[10] = (__bf16)b0.z; r[11] = (__bf16)b0.w;
  r[12] = (__bf16)b1.x; r[13] = (__bf16)b1.y; r[14] = (__bf16)b1.z; r[15] = (__bf16)b1.w;
  return r;
}

// B-fragment (32x16 bf16, KxN): lane holds column n=lane&15; elements are
// 16 consecutive K starting at (lane>=16 ? 16 : 0).
__device__ inline v16bf loadB_bf16(const __bf16* p) {
  BF8 lo, hi;
  lo.u = *reinterpret_cast<const uint4*>(p);
  hi.u = *reinterpret_cast<const uint4*>(p + 8);
  v16bf r;
#pragma unroll
  for (int i = 0; i < 8; ++i) { r[i] = lo.b[i]; r[i + 8] = hi.b[i]; }
  return r;
}

__device__ inline v16bf loadB_f32(const float* p) {
  float4 a = reinterpret_cast<const float4*>(p)[0];
  float4 b = reinterpret_cast<const float4*>(p)[1];
  float4 c = reinterpret_cast<const float4*>(p)[2];
  float4 d = reinterpret_cast<const float4*>(p)[3];
  v16bf r;
  r[0]  = (__bf16)a.x; r[1]  = (__bf16)a.y; r[2]  = (__bf16)a.z; r[3]  = (__bf16)a.w;
  r[4]  = (__bf16)b.x; r[5]  = (__bf16)b.y; r[6]  = (__bf16)b.z; r[7]  = (__bf16)b.w;
  r[8]  = (__bf16)c.x; r[9]  = (__bf16)c.y; r[10] = (__bf16)c.z; r[11] = (__bf16)c.w;
  r[12] = (__bf16)d.x; r[13] = (__bf16)d.y; r[14] = (__bf16)d.z; r[15] = (__bf16)d.w;
  return r;
}

// Tiled GEMM: Out[m, n] = sum_k A[m, k] * W[n, k] + bias[n]
// One wave -> one 64x64 output tile (4 A-frags x 4 B-frags, 16 WMMA / K-step).
// ABF16: A is bf16 (else f32, converted in registers).
// SMODE: 0 = bf16 row-major [M, N]                    (Q/K)
//        1 = bf16 per-head transposed [b, h, dh, s]   (V)
//        2 = f32 row-major [M, N]                     (final output)
template <bool ABF16, int SMODE>
__global__ __launch_bounds__(32) void gemm64_kernel(const void* __restrict__ Ap,
                                                    const float* __restrict__ W,
                                                    const float* __restrict__ bias,
                                                    void* __restrict__ Out) {
  constexpr int Kd = D_, Nd = D_;
  const int mBase = blockIdx.x * 64;
  const int nBase = blockIdx.y * 64;
  const int lane  = threadIdx.x & 31;
  const int half  = lane >> 4;
  const int lm    = lane & 15;
  const int ka    = half * 8;
  const int kboff = half * 16;

  const float*  arow_f[4];
  const __bf16* arow_b[4];
  const float*  wrow[4];
#pragma unroll
  for (int i = 0; i < 4; ++i) {
    arow_f[i] = reinterpret_cast<const float*>(Ap)  + (size_t)(mBase + 16 * i + lm) * Kd;
    arow_b[i] = reinterpret_cast<const __bf16*>(Ap) + (size_t)(mBase + 16 * i + lm) * Kd;
    wrow[i]   = W + (size_t)(nBase + 16 * i + lm) * Kd;
  }

  v8f acc[4][4];
#pragma unroll
  for (int i = 0; i < 4; ++i)
#pragma unroll
    for (int j = 0; j < 4; ++j) acc[i][j] = vzero8();

#pragma unroll 1
  for (int kk = 0; kk < Kd; kk += 32) {
    v16bf a[4], bm[4];
#pragma unroll
    for (int i = 0; i < 4; ++i) {
      if constexpr (ABF16) a[i] = loadA_bf16(arow_b[i] + kk, ka);
      else                 a[i] = loadA_f32(arow_f[i] + kk, ka);
    }
#pragma unroll
    for (int j = 0; j < 4; ++j) bm[j] = loadB_f32(wrow[j] + kk + kboff);
#pragma unroll
    for (int i = 0; i < 4; ++i)
#pragma unroll
      for (int j = 0; j < 4; ++j)
        acc[i][j] = __builtin_amdgcn_wmma_f32_16x16x32_bf16(false, a[i], false, bm[j],
                                                            (short)0, acc[i][j], false, false);
  }

#pragma unroll
  for (int j = 0; j < 4; ++j) {
    const float bv = bias[nBase + 16 * j + lm];
#pragma unroll
    for (int i = 0; i < 4; ++i)
#pragma unroll
      for (int r = 0; r < 8; ++r) acc[i][j][r] += bv;
  }

  if constexpr (SMODE == 0) {
    __bf16* out = reinterpret_cast<__bf16*>(Out);
#pragma unroll
    for (int i = 0; i < 4; ++i)
#pragma unroll
      for (int j = 0; j < 4; ++j)
#pragma unroll
        for (int r = 0; r < 8; ++r)
          out[(size_t)(mBase + 16 * i + r + half * 8) * Nd + nBase + 16 * j + lm] =
              (__bf16)acc[i][j][r];
  } else if constexpr (SMODE == 1) {
    const int bidx = mBase / S_;
    const int s0   = mBase - bidx * S_;
#pragma unroll
    for (int j = 0; j < 4; ++j) {
      const int e  = nBase + 16 * j + lm;
      const int hh = e >> 6;
      const int dh = e & 63;
      __bf16* outc = reinterpret_cast<__bf16*>(Out) +
                     (((size_t)(bidx * H_ + hh) * DH_) + dh) * S_ + s0 + half * 8;
#pragma unroll
      for (int i = 0; i < 4; ++i) {
        BF8 pk;
#pragma unroll
        for (int r = 0; r < 8; ++r) pk.b[r] = (__bf16)acc[i][j][r];
        *reinterpret_cast<uint4*>(outc + 16 * i) = pk.u;  // 8 consecutive s -> 16B store
      }
    }
  } else {
    float* out = reinterpret_cast<float*>(Out);
#pragma unroll
    for (int i = 0; i < 4; ++i)
#pragma unroll
      for (int j = 0; j < 4; ++j)
#pragma unroll
        for (int r = 0; r < 8; ++r)
          out[(size_t)(mBase + 16 * i + r + half * 8) * Nd + nBase + 16 * j + lm] =
              acc[i][j][r];
  }
}

// Flash-attention style causal attention: one wave per (b, h, 32-query group).
// Two 16-row query tiles share every K and V fragment (halves K/V L2 traffic).
// Exactly one 32-key block per group needs masking (qBase0 is a multiple of 32).
// Q, K: bf16 [b, s, d] (d = h*64+dh). Vt: bf16 [b, h, dh, s]. Att: bf16 [b, s, d].
__global__ __launch_bounds__(32) void attn_kernel(const void* __restrict__ Qv,
                                                  const void* __restrict__ Kv,
                                                  const void* __restrict__ Vv,
                                                  void* __restrict__ Av) {
  __shared__ __bf16 pT[2][16 * 32];  // staged P tiles (C-layout -> A-layout)

  const __bf16* Q   = reinterpret_cast<const __bf16*>(Qv);
  const __bf16* Km  = reinterpret_cast<const __bf16*>(Kv);
  const __bf16* Vt  = reinterpret_cast<const __bf16*>(Vv);
  __bf16*       Att = reinterpret_cast<__bf16*>(Av);

  const int qg = blockIdx.x, hh = blockIdx.y, b = blockIdx.z;
  const int lane = threadIdx.x & 31;
  const int half = lane >> 4;
  const int lm   = lane & 15;
  const int ka    = half * 8;
  const int kboff = half * 16;
  const int qBase0 = qg * 32;
  const int qBase[2] = {qBase0, qBase0 + 16};

  // Q fragments, pre-scaled by 1/sqrt(DH) = 0.125 (power of two, exact in bf16)
  v16bf aq[2][2];
#pragma unroll
  for (int qi = 0; qi < 2; ++qi) {
    const __bf16* qrow = Q + ((size_t)(b * S_) + qBase[qi] + lm) * D_ + hh * DH_;
    aq[qi][0] = loadA_bf16(qrow, ka);
    aq[qi][1] = loadA_bf16(qrow + 32, ka);
#pragma unroll
    for (int e = 0; e < 16; ++e) {
      aq[qi][0][e] = (__bf16)((float)aq[qi][0][e] * 0.125f);
      aq[qi][1][e] = (__bf16)((float)aq[qi][1][e] * 0.125f);
    }
  }

  const __bf16* kbase = Km + (size_t)(b * S_) * D_ + hh * DH_;
  const __bf16* vbase = Vt + ((size_t)(b * H_ + hh) * DH_) * S_;

  float rmax[2][8], rsum[2][8];
  v8f oacc[2][4];
#pragma unroll
  for (int qi = 0; qi < 2; ++qi) {
#pragma unroll
    for (int r = 0; r < 8; ++r) { rmax[qi][r] = -__builtin_inff(); rsum[qi][r] = 0.0f; }
#pragma unroll
    for (int j = 0; j < 4; ++j) oacc[qi][j] = vzero8();
  }

  auto doBlock = [&](int kb0, bool masked) {
    // K fragments for this 32-key block, shared by both query tiles
    v16bf kf0[2], kf1[2];
#pragma unroll
    for (int t = 0; t < 2; ++t) {
      const __bf16* krow = kbase + (size_t)(kb0 + t * 16 + lm) * D_;
      kf0[t] = loadB_bf16(krow + kboff);
      kf1[t] = loadB_bf16(krow + 32 + kboff);
    }
#pragma unroll
    for (int qi = 0; qi < 2; ++qi) {
      float p[2][8];
#pragma unroll
      for (int t = 0; t < 2; ++t) {
        v8f sc = vzero8();
        sc = __builtin_amdgcn_wmma_f32_16x16x32_bf16(false, aq[qi][0], false, kf0[t],
                                                     (short)0, sc, false, false);
        sc = __builtin_amdgcn_wmma_f32_16x16x32_bf16(false, aq[qi][1], false, kf1[t],
                                                     (short)0, sc, false, false);
#pragma unroll
        for (int r = 0; r < 8; ++r) p[t][r] = sc[r];
        if (masked) {  // only the diagonal block pays for compares
#pragma unroll
          for (int r = 0; r < 8; ++r) {
            const int qRow = qBase[qi] + r + half * 8;
            const int key  = kb0 + t * 16 + lm;
            if (key > qRow) p[t][r] = -__builtin_inff();
          }
        }
      }
      // online softmax (each row spread across 16 lanes of a half)
#pragma unroll
      for (int r = 0; r < 8; ++r) {
        float bmx = fmaxf(p[0][r], p[1][r]);
#pragma unroll
        for (int off = 1; off < 16; off <<= 1)
          bmx = fmaxf(bmx, __shfl_xor(bmx, off, 32));
        const float mnew = fmaxf(rmax[qi][r], bmx);
        const float mcl  = fmaxf(mnew, -1e30f);
        const float fac  = __expf(rmax[qi][r] - mcl);
        const float p0   = __expf(p[0][r] - mcl);
        const float p1   = __expf(p[1][r] - mcl);
        float bs = p0 + p1;
#pragma unroll
        for (int off = 1; off < 16; off <<= 1)
          bs += __shfl_xor(bs, off, 32);
        rsum[qi][r] = rsum[qi][r] * fac + bs;
        rmax[qi][r] = mnew;
        p[0][r] = p0;
        p[1][r] = p1;
#pragma unroll
        for (int j = 0; j < 4; ++j) oacc[qi][j][r] *= fac;
      }
      // stage P (C-layout) into LDS
#pragma unroll
      for (int t = 0; t < 2; ++t)
#pragma unroll
        for (int r = 0; r < 8; ++r)
          pT[qi][(r + half * 8) * 32 + t * 16 + lm] = (__bf16)p[t][r];
    }
    __syncthreads();
    const v16bf aP0 = loadA_bf16(&pT[0][lm * 32], ka);
    const v16bf aP1 = loadA_bf16(&pT[1][lm * 32], ka);
    // P (16x32) x V (32x64): V fragments shared by both query tiles
#pragma unroll
    for (int j = 0; j < 4; ++j) {
      v16bf bV = loadB_bf16(vbase + (size_t)(j * 16 + lm) * S_ + kb0 + kboff);
      oacc[0][j] = __builtin_amdgcn_wmma_f32_16x16x32_bf16(false, aP0, false, bV,
                                                           (short)0, oacc[0][j], false, false);
      oacc[1][j] = __builtin_amdgcn_wmma_f32_16x16x32_bf16(false, aP1, false, bV,
                                                           (short)0, oacc[1][j], false, false);
    }
    __syncthreads();
  };

  for (int kb0 = 0; kb0 < qBase0; kb0 += 32) doBlock(kb0, false);  // unmasked bulk
  doBlock(qBase0, true);                                           // diagonal block

  // normalize and write attended [b, s, h*64+dh] as bf16
#pragma unroll
  for (int qi = 0; qi < 2; ++qi)
#pragma unroll
    for (int r = 0; r < 8; ++r) {
      const float inv = 1.0f / rsum[qi][r];
      const int row   = qBase[qi] + r + half * 8;
      __bf16* orow = Att + ((size_t)(b * S_) + row) * D_ + hh * DH_;
#pragma unroll
      for (int j = 0; j < 4; ++j)
        orow[j * 16 + lm] = (__bf16)(oacc[qi][j][r] * inv);
    }
}

extern "C" void kernel_launch(void* const* d_in, const int* in_sizes, int n_in,
                              void* d_out, int out_size, void* d_ws, size_t ws_size,
                              hipStream_t stream) {
  (void)in_sizes; (void)n_in; (void)out_size; (void)ws_size;
  const float* x  = (const float*)d_in[0];
  // d_in[1] is the tril mask; causality is computed arithmetically instead.
  const float* Wq = (const float*)d_in[2];
  const float* bq = (const float*)d_in[3];
  const float* Wk = (const float*)d_in[4];
  const float* bk = (const float*)d_in[5];
  const float* Wv = (const float*)d_in[6];
  const float* bv = (const float*)d_in[7];
  const float* Wo = (const float*)d_in[8];
  const float* bo = (const float*)d_in[9];

  // Workspace: Q, K, Vt, Att — each B*S*D bf16 (16.8 MB), total ~67 MB.
  const size_t tensorBytes = (size_t)B_ * S_ * D_ * sizeof(unsigned short);
  char* ws  = (char*)d_ws;
  void* qb  = ws;
  void* kb  = ws + tensorBytes;
  void* vt  = ws + 2 * tensorBytes;
  void* att = ws + 3 * tensorBytes;

  dim3 blk(32);
  dim3 gg(B_ * S_ / 64, D_ / 64);  // 128 x 16 tiles of 64x64
  gemm64_kernel<false, 0><<<gg, blk, 0, stream>>>(x, Wq, bq, qb);
  gemm64_kernel<false, 0><<<gg, blk, 0, stream>>>(x, Wk, bk, kb);
  gemm64_kernel<false, 1><<<gg, blk, 0, stream>>>(x, Wv, bv, vt);

  dim3 ga(S_ / 32, H_, B_);        // 64 x 16 x 4 query groups of 32
  attn_kernel<<<ga, blk, 0, stream>>>(qb, kb, vt, att);

  gemm64_kernel<true, 2><<<gg, blk, 0, stream>>>(att, Wo, bo, d_out);
}
